// SingleQueryAttention_44169443672907
// MI455X (gfx1250) — matrix operand order repaired and checked by reading, not compiled, not run on hardware
//
#include <hip/hip_runtime.h>
#include <hip/hip_bf16.h>

// ---- CDNA5 WMMA vector types ----
typedef __attribute__((ext_vector_type(16))) __bf16 v16bf;
typedef __attribute__((ext_vector_type(8)))  float  v8f;

#define IN_DIM  2048
#define A_DIM   2048
#define O_DIM   2048
#define NB      16
#define LCTX    4096
#define CHUNK   128
#define NCHUNK  32           // LCTX / CHUNK
#define SCALE   0.022097086912079608f   // 1/sqrt(2048)

// Generic 16x16 (M x N) tile GEMM over K=2048 using v_wmma_f32_16x16x32_bf16.
// A element (m,k) at A0[m*am + k]  (A always contiguous in K here -> b128 loads).
// B element (k,n) at B0[k*bks + n*bns]; BVEC=true means bks==1 (contiguous K).
template <bool BVEC>
__device__ __forceinline__ v8f wmma_k2048(const float* __restrict__ A0, int am,
                                          const float* __restrict__ B0, int bks, int bns,
                                          int lane) {
  const int mm = lane & 15;     // A row within tile; also B col (n == lane&15)
  const int h  = lane >> 4;     // lane half selects K sub-range
  const float* Ar = A0 + (size_t)mm * am + 16 * h;
  const float* Bc = B0 + (size_t)mm * bns + (BVEC ? 16 * h : 0);
  v8f c = {};
  for (int kk = 0; kk < 2048; kk += 32) {
    v16bf a, b;
    // A fragment: j -> k = kk + 16h + j
    {
      const float4* ap = (const float4*)(Ar + kk);
      float av[16];
      *(float4*)&av[0]  = ap[0];
      *(float4*)&av[4]  = ap[1];
      *(float4*)&av[8]  = ap[2];
      *(float4*)&av[12] = ap[3];
#pragma unroll
      for (int j = 0; j < 16; ++j) a[j] = (__bf16)av[j];
    }
    // B fragment: j -> k = kk + 16h + j, n = lane&15
    if (BVEC) {
      const float4* bp = (const float4*)(Bc + kk);
      float bv[16];
      *(float4*)&bv[0]  = bp[0];
      *(float4*)&bv[4]  = bp[1];
      *(float4*)&bv[8]  = bp[2];
      *(float4*)&bv[12] = bp[3];
#pragma unroll
      for (int j = 0; j < 16; ++j) b[j] = (__bf16)bv[j];
    } else {
#pragma unroll
      for (int j = 0; j < 16; ++j)
        b[j] = (__bf16)Bc[(size_t)(kk + 16 * h + j) * bks];
    }
    // (neg_a, A, neg_b, B, c_mod, C, reuse_a, reuse_b)
    c = __builtin_amdgcn_wmma_f32_16x16x32_bf16(false, a, false, b, (short)0, c,
                                                false, false);
  }
  return c;
}

// Kernel 1: Q[16,2048] = Xcur[16,2048] @ Wq[2048,2048] + bq   (WMMA)
__global__ __launch_bounds__(128) void sqa_qproj(const float* __restrict__ xc,
                                                 const float* __restrict__ Wq,
                                                 const float* __restrict__ bq,
                                                 float* __restrict__ Q) {
  const int lane = threadIdx.x & 31;
  const int tile = blockIdx.x * 4 + (threadIdx.x >> 5);
  const int n0 = tile * 16;
  v8f c = wmma_k2048<false>(xc, IN_DIM, Wq + n0, A_DIM, 1, lane);
  const int mm = lane & 15, h = lane >> 4;
  const int n = n0 + mm;
#pragma unroll
  for (int r = 0; r < 8; ++r)
    Q[(size_t)(r + 8 * h) * A_DIM + n] = c[r] + bq[n];
}

// Kernel 2: W2[b,i] = scale * sum_a Wk[i,a]*Q[b,a]   (M=2048 rows i, N=16 batches; WMMA)
__global__ __launch_bounds__(128) void sqa_wscore(const float* __restrict__ Wk,
                                                  const float* __restrict__ Q,
                                                  float* __restrict__ W2) {
  const int lane = threadIdx.x & 31;
  const int tile = blockIdx.x * 4 + (threadIdx.x >> 5);
  const int i0 = tile * 16;
  // A = Wk rows i (k = a);  B(k,n) = Q[n*2048 + k]  (transposed access, contiguous K)
  v8f c = wmma_k2048<true>(Wk + (size_t)i0 * A_DIM, A_DIM, Q, 1, A_DIM, lane);
  const int mm = lane & 15, h = lane >> 4;
#pragma unroll
  for (int r = 0; r < 8; ++r)
    W2[(size_t)mm * IN_DIM + (i0 + r + 8 * h)] = c[r] * SCALE;
}

// Kernel 3: single streaming pass over context. Per (batch, 128-row chunk):
// scores = X_row . w_b (softmax-scale already folded into w), chunk-local
// softmax, partial u_c = sum p_l * X_row. Row re-read for u_c hits L2.
__global__ __launch_bounds__(256) void sqa_attn_pass(const float* __restrict__ ctx,
                                                     const float* __restrict__ W2,
                                                     float* __restrict__ uc,
                                                     float* __restrict__ mstat,
                                                     float* __restrict__ sstat) {
  __shared__ float wsh[IN_DIM];
  __shared__ float sc[CHUNK];
  __shared__ float psh[CHUNK];
  const int b = blockIdx.y, cch = blockIdx.x;
  const int tid = threadIdx.x;
  // stage 0: w_b -> LDS (8 KB) via gfx1250 async global->LDS copies (ASYNCcnt)
  {
    const float* src = W2 + (size_t)b * IN_DIM;
#pragma unroll
    for (int i = tid; i < IN_DIM / 4; i += 256) {
      unsigned int       lds_off = (unsigned int)(unsigned long long)(&wsh[i * 4]);
      unsigned long long gaddr   = (unsigned long long)(src + i * 4);
      asm volatile("global_load_async_to_lds_b128 %0, %1, off"
                   :: "v"(lds_off), "v"(gaddr) : "memory");
    }
    asm volatile("s_wait_asynccnt 0x0" ::: "memory");
  }
  __syncthreads();
  const int wave = tid >> 5, lane = tid & 31;
  const size_t rowbase = ((size_t)b * LCTX + (size_t)cch * CHUNK) * IN_DIM;
  // stage A: 16 rows per wave, full 2048-wide dot via float4 + wave reduction
  for (int rr = 0; rr < 16; ++rr) {
    const int l = wave * 16 + rr;
    const float4* xr = (const float4*)(ctx + rowbase + (size_t)l * IN_DIM);
    const float4* wr = (const float4*)wsh;
    if (rr + 1 < 16)
      __builtin_prefetch(ctx + rowbase + (size_t)(l + 1) * IN_DIM + lane * 4, 0, 0);
    float p = 0.f;
#pragma unroll
    for (int s = 0; s < 16; ++s) {
      const float4 x = xr[s * 32 + lane];
      const float4 w = wr[s * 32 + lane];
      p += x.x * w.x + x.y * w.y + x.z * w.z + x.w * w.w;
    }
#pragma unroll
    for (int off = 16; off > 0; off >>= 1) p += __shfl_xor(p, off, 32);
    if (lane == 0) sc[l] = p;
  }
  __syncthreads();
  // stage B: chunk-local max / exp
  float m = -3.4e38f;
  for (int i = 0; i < CHUNK; ++i) m = fmaxf(m, sc[i]);
  if (tid < CHUNK) psh[tid] = __expf(sc[tid] - m);
  __syncthreads();
  const int bc = b * NCHUNK + cch;
  if (wave == 0) {
    float s = 0.f;
    for (int i = lane; i < CHUNK; i += 32) s += psh[i];
#pragma unroll
    for (int off = 16; off > 0; off >>= 1) s += __shfl_xor(s, off, 32);
    if (lane == 0) { mstat[bc] = m; sstat[bc] = s; }
  }
  // stage C: u_c accumulation; each thread owns 8 contiguous columns
  float4 a0 = {0.f, 0.f, 0.f, 0.f}, a1 = {0.f, 0.f, 0.f, 0.f};
  const int i0 = tid * 8;
  for (int l = 0; l < CHUNK; ++l) {
    const float p = psh[l];
    const float4* xr = (const float4*)(ctx + rowbase + (size_t)l * IN_DIM + i0);
    const float4 x0 = xr[0], x1 = xr[1];
    a0.x += p * x0.x; a0.y += p * x0.y; a0.z += p * x0.z; a0.w += p * x0.w;
    a1.x += p * x1.x; a1.y += p * x1.y; a1.z += p * x1.z; a1.w += p * x1.w;
  }
  float4* uo = (float4*)(uc + (size_t)bc * IN_DIM + i0);
  uo[0] = a0; uo[1] = a1;
}

// Kernel 4: cross-chunk log-sum-exp combine -> U[16,2048] (normalized attn @ X)
__global__ __launch_bounds__(256) void sqa_finalize(const float* __restrict__ uc,
                                                    const float* __restrict__ mstat,
                                                    const float* __restrict__ sstat,
                                                    float* __restrict__ U) {
  const int b = blockIdx.x, tid = threadIdx.x;
  float mg = -3.4e38f;
  for (int c = 0; c < NCHUNK; ++c) mg = fmaxf(mg, mstat[b * NCHUNK + c]);
  float S = 0.f;
  float4 a0 = {0.f, 0.f, 0.f, 0.f}, a1 = {0.f, 0.f, 0.f, 0.f};
  const int i0 = tid * 8;
  for (int c = 0; c < NCHUNK; ++c) {
    const float w = __expf(mstat[b * NCHUNK + c] - mg);
    S += sstat[b * NCHUNK + c] * w;
    const float4* up = (const float4*)(uc + (size_t)(b * NCHUNK + c) * IN_DIM + i0);
    const float4 u0 = up[0], u1 = up[1];
    a0.x += w * u0.x; a0.y += w * u0.y; a0.z += w * u0.z; a0.w += w * u0.w;
    a1.x += w * u1.x; a1.y += w * u1.y; a1.z += w * u1.z; a1.w += w * u1.w;
  }
  const float inv = 1.f / S;
  a0.x *= inv; a0.y *= inv; a0.z *= inv; a0.w *= inv;
  a1.x *= inv; a1.y *= inv; a1.z *= inv; a1.w *= inv;
  float4* uo = (float4*)(U + (size_t)b * IN_DIM + i0);
  uo[0] = a0; uo[1] = a1;
}

// Kernel 5: out[16,2048] = U @ Wv + bv   (WMMA; +bv valid since softmax sums to 1)
__global__ __launch_bounds__(128) void sqa_outproj(const float* __restrict__ U,
                                                   const float* __restrict__ Wv,
                                                   const float* __restrict__ bv,
                                                   float* __restrict__ out) {
  const int lane = threadIdx.x & 31;
  const int tile = blockIdx.x * 4 + (threadIdx.x >> 5);
  const int n0 = tile * 16;
  v8f c = wmma_k2048<false>(U, IN_DIM, Wv + n0, O_DIM, 1, lane);
  const int mm = lane & 15, h = lane >> 4;
  const int n = n0 + mm;
#pragma unroll
  for (int r = 0; r < 8; ++r)
    out[(size_t)(r + 8 * h) * O_DIM + n] = c[r] + bv[n];
}

extern "C" void kernel_launch(void* const* d_in, const int* in_sizes, int n_in,
                              void* d_out, int out_size, void* d_ws, size_t ws_size,
                              hipStream_t stream) {
  (void)in_sizes; (void)n_in; (void)out_size; (void)ws_size;
  const float* xc  = (const float*)d_in[0];  // current_token [16,1,2048]
  const float* ctx = (const float*)d_in[1];  // context_tokens [16,4096,2048]
  const float* Wq  = (const float*)d_in[2];
  const float* bq  = (const float*)d_in[3];
  const float* Wk  = (const float*)d_in[4];
  // bk (d_in[5]) adds a per-batch constant to all scores -> softmax invariant; dropped.
  const float* Wv  = (const float*)d_in[6];
  const float* bv  = (const float*)d_in[7];

  float* ws    = (float*)d_ws;
  float* Q     = ws;                       // 16*2048
  float* W2    = ws + 32768;               // 16*2048  (Wk @ q, scaled)
  float* U     = ws + 65536;               // 16*2048
  float* mstat = ws + 98304;               // 512
  float* sstat = ws + 98816;               // 512
  float* uc    = ws + 102400;              // 512 * 2048 partial u vectors

  sqa_qproj <<<32, 128, 0, stream>>>(xc, Wq, bq, Q);
  sqa_wscore<<<32, 128, 0, stream>>>(Wk, Q, W2);
  sqa_attn_pass<<<dim3(NCHUNK, NB), 256, 0, stream>>>(ctx, W2, uc, mstat, sstat);
  sqa_finalize<<<NB, 256, 0, stream>>>(uc, mstat, sstat, U);
  sqa_outproj<<<32, 128, 0, stream>>>(U, Wv, bv, (float*)d_out);
}
